// UnifiedQuantizedLinear_70506183131205
// MI455X (gfx1250) — compile-verified, matrix operand order checked by reading, not run on hardware
//
#include <hip/hip_runtime.h>

// ---------------------------------------------------------------------------
// NF4-quantized linear  y = x @ W^T + bias   (MI455X / gfx1250, wave32, WMMA)
//   x:      [32, 1, 4096]  f32
//   packed: [11008*4096/2] int32, one packed byte per int32 (lo nib = even k)
//   scales: [11008*4096/128] f32 (group = 128 consecutive k within a row)
//   bias:   [11008] f32
//   out:    [32, 1, 11008] f32
// Memory-bound: 90 MB of packed weights read exactly once; compute via
// v_wmma_f32_16x16x32_f16 (dequant NF4 -> f16 in registers via LDS table).
// ---------------------------------------------------------------------------

typedef __attribute__((ext_vector_type(16))) _Float16 v16h;
typedef __attribute__((ext_vector_type(8)))  _Float16 v8h;
typedef __attribute__((ext_vector_type(2)))  _Float16 h2;
typedef __attribute__((ext_vector_type(8)))  float    v8f;
typedef __attribute__((ext_vector_type(4)))  int      v4i;
typedef __attribute__((ext_vector_type(4)))  float    v4f;

#define IN_F        4096
#define OUT_F       11008
#define GROUP_SZ    128
#define GROUPS_ROW  (IN_F / GROUP_SZ)      // 32
#define M_BATCH     32
#define KSPLIT      8
#define KCHUNK      (IN_F / KSPLIT)        // 512
#define KSTEPS      (KCHUNK / 32)          // 16
#define OUT_ELEMS   (M_BATCH * OUT_F)      // 352256  (per-partial size, floats)

__device__ __constant__ float NF4_LEVELS_D[16] = {
    -1.0f, -0.6961928009986877f, -0.5250730514526367f, -0.39491748809814453f,
    -0.28444138169288635f, -0.18477343022823334f, -0.09105003625154495f, 0.0f,
    0.07958029955625534f, 0.16093020141124725f, 0.24611230194568634f,
    0.33791524171829224f, 0.44070982933044434f, 0.5626170039176941f,
    0.7229568362236023f, 1.0f};

__device__ __forceinline__ h2 deq_pair(const unsigned int* tab, int code, h2 s2) {
    // tab[code] = {NF4[lo], NF4[hi]} as packed f16; scale by group absmax
    return __builtin_bit_cast(h2, tab[code & 255]) * s2;
}

// One wave computes a 32(M) x 16(N) f32 tile over K in [kz*512, kz*512+512).
// Block = 128 threads = 4 waves = 64 output columns.  Grid = (172, 8).
__global__ __launch_bounds__(128)
void nf4_wmma_gemm(const float* __restrict__ x,
                   const int*   __restrict__ packed,
                   const float* __restrict__ scales,
                   float*       __restrict__ part) {
    __shared__ _Float16     lds_x[M_BATCH * KCHUNK];  // 32 KB, x chunk as f16
    __shared__ unsigned int lds_tab[256];             // 1 KB, byte -> half2 levels

    const int tid   = threadIdx.x;
    const int kz    = blockIdx.y;
    const int kbase = kz * KCHUNK;

    // --- build byte->pair(level) dequant table -----------------------------
    for (int e = tid; e < 256; e += 128) {
        h2 v;
        v.x = (_Float16)NF4_LEVELS_D[e & 15];         // low nibble  = even k
        v.y = (_Float16)NF4_LEVELS_D[(e >> 4) & 15];  // high nibble = odd  k
        lds_tab[e] = __builtin_bit_cast(unsigned int, v);
    }

    // --- stage x[0:32, kbase:kbase+512] into LDS as f16 --------------------
    for (int i = tid * 4; i < M_BATCH * KCHUNK; i += 128 * 4) {
        const int row = i >> 9;             // / KCHUNK
        const int col = i & (KCHUNK - 1);
        v4f xv = *(const v4f*)(x + row * IN_F + kbase + col);
        h2 p0; p0.x = (_Float16)xv.x; p0.y = (_Float16)xv.y;
        h2 p1; p1.x = (_Float16)xv.z; p1.y = (_Float16)xv.w;
        *(h2*)(&lds_x[i])     = p0;
        *(h2*)(&lds_x[i + 2]) = p1;
    }
    __syncthreads();

    const int wave  = tid >> 5;
    const int lane  = tid & 31;
    const int col   = lane & 15;   // N within tile; also A-row index
    const int khalf = lane >> 4;   // which K half this lane holds (WMMA layout)
    const int n     = blockIdx.x * 64 + wave * 16 + col;

    v8f c0 = {};  // rows M=0..15
    v8f c1 = {};  // rows M=16..31

    const int prow = n * (IN_F / 2);        // packed byte index of row n start

    for (int step = 0; step < KSTEPS; ++step) {
        const int k0 = kbase + step * 32;

        // ---- B fragment: 32(K) x 16(N) f16, dequantized on the fly -------
        // lane holds k offsets {8*khalf+0..7} and {16+8*khalf+0..7}
        const int bidx = prow + (k0 >> 1) + khalf * 4;
        v4i pk0 = *(const v4i*)(packed + bidx);       // bytes for K 8h+0..7
        v4i pk1 = *(const v4i*)(packed + bidx + 8);   // bytes for K 16+8h+0..7
        if (step + 1 < KSTEPS)                        // prefetch next step
            __builtin_prefetch((const void*)(packed + bidx + 16), 0, 0);

        const float s = scales[n * GROUPS_ROW + (k0 >> 7)];
        h2 s2; s2.x = (_Float16)s; s2.y = (_Float16)s;

        h2 w0 = deq_pair(lds_tab, pk0.x, s2);
        h2 w1 = deq_pair(lds_tab, pk0.y, s2);
        h2 w2 = deq_pair(lds_tab, pk0.z, s2);
        h2 w3 = deq_pair(lds_tab, pk0.w, s2);
        h2 w4 = deq_pair(lds_tab, pk1.x, s2);
        h2 w5 = deq_pair(lds_tab, pk1.y, s2);
        h2 w6 = deq_pair(lds_tab, pk1.z, s2);
        h2 w7 = deq_pair(lds_tab, pk1.w, s2);

        v16h b;
        b[0]  = w0.x; b[1]  = w0.y; b[2]  = w1.x; b[3]  = w1.y;
        b[4]  = w2.x; b[5]  = w2.y; b[6]  = w3.x; b[7]  = w3.y;
        b[8]  = w4.x; b[9]  = w4.y; b[10] = w5.x; b[11] = w5.y;
        b[12] = w6.x; b[13] = w6.y; b[14] = w7.x; b[15] = w7.y;

        // ---- A fragments from LDS (16-bit A layout: contiguous K octets) --
        const int abase = step * 32 + khalf * 8;
        const _Float16* ap0 = &lds_x[col * KCHUNK + abase];
        const _Float16* ap1 = &lds_x[(col + 16) * KCHUNK + abase];
        v8h a0l = *(const v8h*)ap0;
        v8h a0h = *(const v8h*)(ap0 + 16);
        v8h a1l = *(const v8h*)ap1;
        v8h a1h = *(const v8h*)(ap1 + 16);
        v16h a0 = __builtin_shufflevector(a0l, a0h, 0,1,2,3,4,5,6,7,8,9,10,11,12,13,14,15);
        v16h a1 = __builtin_shufflevector(a1l, a1h, 0,1,2,3,4,5,6,7,8,9,10,11,12,13,14,15);

        // ---- D = A x B + C ------------------------------------------------
        c0 = __builtin_amdgcn_wmma_f32_16x16x32_f16(false, a0, false, b,
                                                    (short)0, c0, false, false);
        c1 = __builtin_amdgcn_wmma_f32_16x16x32_f16(false, a1, false, b,
                                                    (short)0, c1, false, false);
    }

    // ---- write this K-split's partial tile (every element rewritten) -----
    float* p = part + kz * OUT_ELEMS;
    const int mrow = khalf * 8;   // C layout: VGPR r -> M = r + 8*(lane>=16)
#pragma unroll
    for (int r = 0; r < 8; ++r) {
        p[(mrow + r) * OUT_F + n]        = c0[r];
        p[(mrow + r + 16) * OUT_F + n]   = c1[r];
    }
}

// Deterministic reduction: out = bias + sum of 8 K-split partials.
__global__ __launch_bounds__(256)
void nf4_reduce_bias(const float* __restrict__ part,
                     const float* __restrict__ bias,
                     float*       __restrict__ out) {
    const int nidx = blockIdx.x * 256 + threadIdx.x;   // 0..11007 (43*256)
    const int m    = blockIdx.y;                       // 0..31
    const int idx  = m * OUT_F + nidx;
    float acc = bias[nidx];
#pragma unroll
    for (int z = 0; z < KSPLIT; ++z)
        acc += part[z * OUT_ELEMS + idx];
    out[idx] = acc;
}

extern "C" void kernel_launch(void* const* d_in, const int* in_sizes, int n_in,
                              void* d_out, int out_size, void* d_ws, size_t ws_size,
                              hipStream_t stream) {
    const float* x      = (const float*)d_in[0];
    const int*   packed = (const int*)d_in[1];
    const float* scales = (const float*)d_in[2];
    const float* bias   = (const float*)d_in[3];
    float*       out    = (float*)d_out;
    float*       part   = (float*)d_ws;   // needs KSPLIT*352256*4 ~= 10.8 MB

    // GEMM: grid (11008/64 = 172 column-blocks, 8 K-splits), 4 waves/block
    nf4_wmma_gemm<<<dim3(OUT_F / 64, KSPLIT), 128, 0, stream>>>(x, packed, scales, part);
    // Reduce partials + bias -> out
    nf4_reduce_bias<<<dim3(OUT_F / 256, M_BATCH), 256, 0, stream>>>(part, bias, out);
}
